// GINEncoderBlock_19885698580759
// MI455X (gfx1250) — matrix operand order repaired and checked by reading, not compiled, not run on hardware
//
#include <hip/hip_runtime.h>
#include <hip/hip_bf16.h>

#define Hc   128
#define H2c  256
#define BN_EPS 1e-5f

typedef __bf16 bf16_t;
typedef bf16_t v16bf __attribute__((ext_vector_type(16)));
typedef bf16_t v8bf  __attribute__((ext_vector_type(8)));
typedef float  v8f   __attribute__((ext_vector_type(8)));
typedef float  v4f   __attribute__((ext_vector_type(4)));

// ---------------------------------------------------------------------------
// Kernel 0: zero agg + BN sums; convert W1/W2 to transposed bf16 in workspace.
//   W1t[n*128 + k] = bf16(W1[k*256 + n])   (n: 0..255 output col, k: 0..127)
//   W2t[n*256 + k] = bf16(W2[k*128 + n])   (n: 0..127 output col, k: 0..255)
// ---------------------------------------------------------------------------
__global__ void init_kernel(const float* __restrict__ W1, const float* __restrict__ W2,
                            float* __restrict__ agg, float* __restrict__ sums,
                            bf16_t* __restrict__ W1t, bf16_t* __restrict__ W2t,
                            long long NH) {
  long long i = (long long)blockIdx.x * blockDim.x + threadIdx.x;
  if (i < NH) agg[i] = 0.0f;
  if (i < 256) sums[i] = 0.0f;                 // [0..127]=sum, [128..255]=sumsq
  if (i < Hc * H2c) {                          // 32768: W1 transpose
    int n = (int)(i >> 7);                     // 0..255
    int k = (int)(i & 127);                    // 0..127
    W1t[n * Hc + k] = (bf16_t)W1[k * H2c + n];
  }
  if (i < H2c * Hc) {                          // 32768: W2 transpose
    int n = (int)(i / H2c);                    // 0..127
    int k = (int)(i % H2c);                    // 0..255
    W2t[n * H2c + k] = (bf16_t)W2[k * Hc + n];
  }
}

// ---------------------------------------------------------------------------
// Kernel 1: per-edge message + scatter-add.
// 32 lanes per edge, 4 channels per lane (covers H=128).
// msg = relu(x[src] + edge_attr@We + be); atomicAdd into agg[dst].
// ---------------------------------------------------------------------------
__global__ __launch_bounds__(256)
void edge_kernel(const float* __restrict__ x,
                 const long long* __restrict__ eidx,   // [2, E]
                 const float* __restrict__ ea,         // [E, 16]
                 const float* __restrict__ We,         // [16, 128]
                 const float* __restrict__ be,         // [128]
                 float* __restrict__ agg, int E_) {
  int t = threadIdx.x;
  int lane = t & 31;
  int e = blockIdx.x * 8 + (t >> 5);
  if (e >= E_) return;
  // prefetch next block's stream data (edge_attr + edge_index are the only
  // true HBM streams; x/agg are L2-resident) -> global_prefetch_b8
  {
    int ep = e + 8;
    if (ep < E_) {
      __builtin_prefetch(ea + (long long)ep * 16, 0, 1);
      __builtin_prefetch(eidx + ep, 0, 1);
      __builtin_prefetch(eidx + (long long)E_ + ep, 0, 1);
    }
  }
  long long src = eidx[e];
  long long dst = eidx[(long long)E_ + e];
  int c = lane * 4;
  const v4f* We4 = (const v4f*)We;                       // [16][32] of float4
  const v4f* ea4 = (const v4f*)(ea + (long long)e * 16); // 4 x float4
  v4f acc = *(const v4f*)(be + c);
  #pragma unroll
  for (int kk = 0; kk < 4; ++kk) {
    v4f a = ea4[kk];
    #pragma unroll
    for (int j = 0; j < 4; ++j) {
      v4f w = We4[(kk * 4 + j) * 32 + lane];
      float s = a[j];
      acc[0] = fmaf(s, w[0], acc[0]);
      acc[1] = fmaf(s, w[1], acc[1]);
      acc[2] = fmaf(s, w[2], acc[2]);
      acc[3] = fmaf(s, w[3], acc[3]);
    }
  }
  v4f xv = *(const v4f*)(x + src * Hc + c);
  float* aggp = agg + dst * Hc + c;
  #pragma unroll
  for (int j = 0; j < 4; ++j) {
    float m = fmaxf(acc[j] + xv[j], 0.0f);
    atomicAdd(aggp + j, m);                               // global_atomic_add_f32
  }
}

// ---------------------------------------------------------------------------
// helpers: fragment loads
// ---------------------------------------------------------------------------
__device__ __forceinline__ v16bf load_b_frag(const bf16_t* __restrict__ p) {
  // bf16 {p..p+7} and {p+16..p+23} -> two global_load_b128
  const v8bf* bp = (const v8bf*)p;
  v8bf bl = bp[0], bh = bp[2];
  v16bf f;
  #pragma unroll
  for (int q = 0; q < 8; ++q) { f[q] = bl[q]; f[q + 8] = bh[q]; }
  return f;
}

__device__ __forceinline__ v16bf load_a_frag(const bf16_t* row, int kb, int koffA) {
  // A frag: K in {kb+koffA .. +7} and {kb+16+koffA .. +7} -> two ds_load_b128
  v8bf al = *(const v8bf*)(row + kb + koffA);
  v8bf ah = *(const v8bf*)(row + kb + 16 + koffA);
  v16bf f;
  #pragma unroll
  for (int q = 0; q < 8; ++q) { f[q] = al[q]; f[q + 8] = ah[q]; }
  return f;
}

// ---------------------------------------------------------------------------
// Kernel 2: node MLP via WMMA bf16.
// Block = 256 threads (8 waves) handles 64 node rows.
//   stage1: h0 = (1+eps)*x + agg  -> bf16 LDS [64][136]
//   stage2: t = relu(h0@W1 + b1)  -> bf16 LDS [64][264]   (WMMA 16x16x32 bf16, K=128)
//   stage3: h = t@W2 + b2         -> hbuf f32, + column sum/sumsq atomics (K=256)
// B (weight) fragments are hoisted into registers outside the row-tile loop.
// ---------------------------------------------------------------------------
__global__ __launch_bounds__(256)
void mlp_kernel(const float* __restrict__ x, const float* __restrict__ agg,
                const float* __restrict__ eps_p,
                const bf16_t* __restrict__ W1t, const float* __restrict__ b1,
                const bf16_t* __restrict__ W2t, const float* __restrict__ b2,
                float* __restrict__ hbuf, float* __restrict__ sums, int N_) {
  __shared__ bf16_t sH0[64][136];   // 17408 B (row stride 272B = 16B-aligned)
  __shared__ bf16_t sT[64][264];    // 33792 B (row stride 528B = 16B-aligned)
  int tid  = threadIdx.x;
  int lane = tid & 31;
  int wave = tid >> 5;
  int r0 = blockIdx.x * 64;
  float epsv = 1.0f + eps_p[0];

  // ---- stage 1: load + bf16 convert ----
  #pragma unroll
  for (int i = 0; i < 8; ++i) {
    int g = tid + i * 256;          // float4 group id, 0..2047
    int row = g >> 5;               // 32 groups per 128-wide row
    int col = (g & 31) * 4;
    v4f h;
    if (r0 + row < N_) {
      long long off = (long long)(r0 + row) * Hc + col;
      v4f xv = *(const v4f*)(x + off);
      v4f av = *(const v4f*)(agg + off);
      h[0] = fmaf(epsv, xv[0], av[0]);
      h[1] = fmaf(epsv, xv[1], av[1]);
      h[2] = fmaf(epsv, xv[2], av[2]);
      h[3] = fmaf(epsv, xv[3], av[3]);
    } else {
      h[0] = h[1] = h[2] = h[3] = 0.0f;
    }
    sH0[row][col + 0] = (bf16_t)h[0];
    sH0[row][col + 1] = (bf16_t)h[1];
    sH0[row][col + 2] = (bf16_t)h[2];
    sH0[row][col + 3] = (bf16_t)h[3];
  }
  __syncthreads();

  int halfsel = lane >> 4;          // 0: lanes 0-15, 1: lanes 16-31
  int nlo     = lane & 15;
  int koffA   = halfsel * 8;        // A frag: K pattern {0..7,16..23} (+8 for hi lanes)
  int koffB   = halfsel * 8;        // B frag: K pattern {0..7,16..23} (+8 for hi lanes)

  // ---- stage 2: GEMM1 [64x128]@[128x256] ----
  {
    int n0 = (wave * 2 + 0) * 16 + nlo;
    int n1 = (wave * 2 + 1) * 16 + nlo;
    float bias0 = b1[n0];
    float bias1 = b1[n1];
    // preload all B fragments for this wave's two column tiles (K=128 -> 4 each)
    v16bf bf0[4], bf1[4];
    #pragma unroll
    for (int kbi = 0; kbi < 4; ++kbi) {
      bf0[kbi] = load_b_frag(W1t + (size_t)n0 * Hc + kbi * 32 + koffB);
      bf1[kbi] = load_b_frag(W1t + (size_t)n1 * Hc + kbi * 32 + koffB);
    }
    for (int mt = 0; mt < 4; ++mt) {                      // 4 row-tiles
      int m = mt * 16 + nlo;                              // A row for this lane
      v16bf af[4];
      #pragma unroll
      for (int kbi = 0; kbi < 4; ++kbi)
        af[kbi] = load_a_frag(&sH0[m][0], kbi * 32, koffA);
      v8f acc0, acc1;
      #pragma unroll
      for (int r = 0; r < 8; ++r) { acc0[r] = bias0; acc1[r] = bias1; }
      #pragma unroll
      for (int kbi = 0; kbi < 4; ++kbi)
        acc0 = __builtin_amdgcn_wmma_f32_16x16x32_bf16(
            false, af[kbi], false, bf0[kbi], (short)0, acc0, false, false);
      #pragma unroll
      for (int kbi = 0; kbi < 4; ++kbi)
        acc1 = __builtin_amdgcn_wmma_f32_16x16x32_bf16(
            false, af[kbi], false, bf1[kbi], (short)0, acc1, false, false);
      #pragma unroll
      for (int r = 0; r < 8; ++r) {
        int mm = mt * 16 + r + halfsel * 8;               // D row mapping
        sT[mm][(wave * 2 + 0) * 16 + nlo] = (bf16_t)fmaxf(acc0[r], 0.0f);
        sT[mm][(wave * 2 + 1) * 16 + nlo] = (bf16_t)fmaxf(acc1[r], 0.0f);
      }
    }
  }
  __syncthreads();

  // ---- stage 3: GEMM2 [64x256]@[256x128] + BN partial sums ----
  {
    int n = wave * 16 + nlo;                              // 1 col-tile per wave
    float bias = b2[n];
    // preload all 8 B fragments (K=256)
    v16bf bf2[8];
    #pragma unroll
    for (int kbi = 0; kbi < 8; ++kbi)
      bf2[kbi] = load_b_frag(W2t + (size_t)n * H2c + kbi * 32 + koffB);
    float lsum = 0.0f, lsq = 0.0f;
    for (int mt = 0; mt < 4; ++mt) {
      int m = mt * 16 + nlo;
      v8f acc;
      #pragma unroll
      for (int r = 0; r < 8; ++r) acc[r] = bias;
      #pragma unroll
      for (int kbi = 0; kbi < 8; ++kbi) {
        v16bf af = load_a_frag(&sT[m][0], kbi * 32, koffA);
        acc = __builtin_amdgcn_wmma_f32_16x16x32_bf16(
            false, af, false, bf2[kbi], (short)0, acc, false, false);
      }
      #pragma unroll
      for (int r = 0; r < 8; ++r) {
        int mm = mt * 16 + r + halfsel * 8;
        int g = r0 + mm;
        if (g < N_) {
          float v = acc[r];
          hbuf[(long long)g * Hc + n] = v;
          lsum += v;
          lsq  += v * v;
        }
      }
    }
    atomicAdd(&sums[n], lsum);
    atomicAdd(&sums[128 + n], lsq);
  }
}

// ---------------------------------------------------------------------------
// Kernel 3: BatchNorm (biased var) + residual + ReLU.
// ---------------------------------------------------------------------------
__global__ __launch_bounds__(256)
void bn_final_kernel(const float* __restrict__ hbuf, const float* __restrict__ x,
                     const float* __restrict__ sums,
                     const float* __restrict__ gamma, const float* __restrict__ beta,
                     float* __restrict__ out, int N_) {
  long long i = ((long long)blockIdx.x * blockDim.x + threadIdx.x) * 4;
  long long total = (long long)N_ * Hc;
  if (i >= total) return;
  int c = (int)(i & (Hc - 1));
  float invN = 1.0f / (float)N_;
  v4f h  = *(const v4f*)(hbuf + i);
  v4f xv = *(const v4f*)(x + i);
  v4f o;
  #pragma unroll
  for (int j = 0; j < 4; ++j) {
    int cc = c + j;
    float mean = sums[cc] * invN;
    float var  = sums[128 + cc] * invN - mean * mean;
    float rstd = rsqrtf(var + BN_EPS);
    float v = (h[j] - mean) * rstd * gamma[cc] + beta[cc];
    o[j] = fmaxf(v + xv[j], 0.0f);
  }
  *(v4f*)(out + i) = o;
}

// ---------------------------------------------------------------------------
extern "C" void kernel_launch(void* const* d_in, const int* in_sizes, int n_in,
                              void* d_out, int out_size, void* d_ws, size_t ws_size,
                              hipStream_t stream) {
  const float*     x     = (const float*)d_in[0];
  const long long* eidx  = (const long long*)d_in[1];
  const float*     ea    = (const float*)d_in[2];
  const float*     We    = (const float*)d_in[3];
  const float*     be    = (const float*)d_in[4];
  const float*     eps   = (const float*)d_in[5];
  const float*     W1    = (const float*)d_in[6];
  const float*     b1    = (const float*)d_in[7];
  const float*     W2    = (const float*)d_in[8];
  const float*     b2    = (const float*)d_in[9];
  const float*     gamma = (const float*)d_in[10];
  const float*     beta  = (const float*)d_in[11];
  float*           out   = (float*)d_out;

  int N_ = in_sizes[0] / Hc;        // 50000
  int E_ = in_sizes[1] / 2;         // 600000
  long long NH = (long long)N_ * Hc;

  // workspace layout
  float*  agg  = (float*)d_ws;               // NH f32
  float*  hbuf = agg + NH;                   // NH f32
  float*  sums = hbuf + NH;                  // 256 f32 (sum | sumsq)
  bf16_t* W1t  = (bf16_t*)(sums + 256);      // 256*128 bf16
  bf16_t* W2t  = W1t + Hc * H2c;             // 128*256 bf16

  {
    int blocks = (int)((NH + 255) / 256);
    init_kernel<<<blocks, 256, 0, stream>>>(W1, W2, agg, sums, W1t, W2t, NH);
  }
  {
    int blocks = (E_ + 7) / 8;
    edge_kernel<<<blocks, 256, 0, stream>>>(x, eidx, ea, We, be, agg, E_);
  }
  {
    int blocks = (N_ + 63) / 64;
    mlp_kernel<<<blocks, 256, 0, stream>>>(x, agg, eps, W1t, b1, W2t, b2, hbuf, sums, N_);
  }
  {
    long long groups = NH / 4;
    int blocks = (int)((groups + 255) / 256);
    bn_final_kernel<<<blocks, 256, 0, stream>>>(hbuf, x, sums, gamma, beta, out, N_);
  }
}
